// AttentionBlock_14259291422857
// MI455X (gfx1250) — matrix operand order repaired and checked by reading, not compiled, not run on hardware
//
#include <hip/hip_runtime.h>
#include <hip/hip_bf16.h>

typedef __attribute__((ext_vector_type(16))) _Float16 v16h;
typedef __attribute__((ext_vector_type(8)))  float    v8f;
typedef __attribute__((ext_vector_type(4)))  int      v4i;

constexpr int  B_  = 16;
constexpr int  C_  = 512;
constexpr int  L_  = 1024;
constexpr int  EC_ = 512;
constexpr int  LE_ = 128;
constexpr int  H_  = 8;
constexpr int  G_  = 32;
constexpr int  CH_ = 64;            // C/H
constexpr int  S_  = LE_ + L_;      // 1152 (concat ek|k)
constexpr float EPS_ = 1e-5f;
constexpr float SCL_ = 0.35355339059327373f;  // 1/sqrt(sqrt(64))

#if __has_builtin(__builtin_amdgcn_global_load_async_to_lds_b128) && \
    __has_builtin(__builtin_amdgcn_s_wait_asynccnt)
#define HAVE_ASYNC_LDS 1
#else
#define HAVE_ASYNC_LDS 0
#endif

#if HAVE_ASYNC_LDS
// builtin signature (from clang diagnostic): (v4i as1*, v4i as3*, Ii, Ii)
typedef __attribute__((address_space(1))) v4i* gas_v4;
typedef __attribute__((address_space(3))) v4i* las_v4;
#endif

__device__ __forceinline__ v8f wmma16(v16h a, v16h b, v8f c) {
  // D = A(16x32 f16) * B(32x16 f16) + C(16x16 f32)
  return __builtin_amdgcn_wmma_f32_16x16x32_f16(false, a, false, b,
                                                (short)0, c, false, false);
}

// ---------------------------------------------------------------------------
// Kernel 1: GroupNorm, fp32 -> f16, store transposed h_T[b][l][c]
// One block per (b, g): 16 channels x 1024 elements.
// ---------------------------------------------------------------------------
__global__ void __launch_bounds__(256)
gn_kernel(const float* __restrict__ x, const float* __restrict__ gscale,
          const float* __restrict__ gbias, _Float16* __restrict__ hT) {
  const int bg = blockIdx.x;
  const int b  = bg / G_;
  const int g  = bg % G_;
  const int cg = C_ / G_;           // 16 channels per group
  const int c0 = g * cg;
  const int n  = cg * L_;           // 16384 elements per group
  const float* xp = x + ((size_t)b * C_ + c0) * L_;

  float s = 0.f, ss = 0.f;
  for (int i = threadIdx.x; i < n; i += blockDim.x) {
    float v = xp[i];
    s += v; ss += v * v;
  }
  #pragma unroll
  for (int m = 16; m > 0; m >>= 1) {
    s  += __shfl_xor(s,  m, 32);
    ss += __shfl_xor(ss, m, 32);
  }
  __shared__ float smem[16];
  const int wid = threadIdx.x >> 5, lid = threadIdx.x & 31;
  if (lid == 0) { smem[wid] = s; smem[8 + wid] = ss; }
  __syncthreads();
  if (threadIdx.x == 0) {
    float S = 0.f, SS = 0.f;
    #pragma unroll
    for (int i = 0; i < 8; i++) { S += smem[i]; SS += smem[8 + i]; }
    float mean = S / (float)n;
    float var  = SS / (float)n - mean * mean;
    smem[0] = mean;
    smem[1] = rsqrtf(var + EPS_);
  }
  __syncthreads();
  const float mean = smem[0], inv = smem[1];
  for (int i = threadIdx.x; i < n; i += blockDim.x) {
    int cc = i / L_, l = i % L_;
    int c  = c0 + cc;
    float v = (xp[i] - mean) * inv * gscale[c] + gbias[c];
    hT[((size_t)b * L_ + l) * C_ + c] = (_Float16)v;
  }
}

// ---------------------------------------------------------------------------
// Shared GEMM core: acc[4] (16x64 tile) = W(f32)[M0..+16, :K] x Bmat(f16).
// B fragments are 32B/lane contiguous; on CDNA5 they are staged through LDS
// with GLOBAL_LOAD_ASYNC_TO_LDS_B128 double-buffering (ASYNCcnt), so the
// WMMAs of step i overlap the global traffic of step i+1.
// ---------------------------------------------------------------------------
#define GEMM_CORE(KDIM, ACC, WPTR, BPTR)                                       \
  do {                                                                         \
    _Float16* myB = &ldsB[wave][0][0];                                         \
    ASYNC_ISSUE(0, 0);                                                         \
    int buf = 0;                                                               \
    _Pragma("unroll 1")                                                        \
    for (int K0 = 0; K0 < (KDIM); K0 += 32) {                                  \
      v16h afrag;                                                              \
      const float* wr = (WPTR) + (size_t)(M0 + col) * (KDIM) + K0 + hi * 16;   \
      __builtin_prefetch(wr + 32, 0, 1);                                       \
      _Pragma("unroll")                                                        \
      for (int j = 0; j < 8; j++) {                                            \
        afrag[j]     = (_Float16)wr[j];                                        \
        afrag[8 + j] = (_Float16)wr[8 + j];                                    \
      }                                                                        \
      if (K0 + 32 < (KDIM)) {                                                  \
        ASYNC_ISSUE(K0 + 32, buf ^ 1);                                         \
        ASYNC_WAIT(8);                                                         \
      } else {                                                                 \
        ASYNC_WAIT(0);                                                         \
      }                                                                        \
      _Pragma("unroll")                                                        \
      for (int nt = 0; nt < 4; nt++) {                                         \
        v16h bb = ASYNC_FRAG(BPTR, KDIM);                                      \
        (ACC)[nt] = wmma16(afrag, bb, (ACC)[nt]);                              \
      }                                                                        \
      buf ^= 1;                                                                \
    }                                                                          \
  } while (0)

#if HAVE_ASYNC_LDS
// issue 8 async b128s: 4 B tiles, 32B per lane, into buffer `bf`
#define ASYNC_DECL __shared__ __align__(32) _Float16 ldsB[4][2][2048]
#define ASYNC_ISSUE(K0v, bf)                                                   \
  do {                                                                         \
    _Pragma("unroll")                                                          \
    for (int nt = 0; nt < 4; nt++) {                                           \
      const _Float16* gp =                                                     \
          Bbase + (size_t)(N0 + nt * 16 + col) * Kdim_ + (K0v) + hi * 16;      \
      _Float16* lp = &ldsB[wave][bf][nt * 512 + lane * 16];                    \
      __builtin_amdgcn_global_load_async_to_lds_b128((gas_v4)gp, (las_v4)lp,   \
                                                     0, 0);                    \
      __builtin_amdgcn_global_load_async_to_lds_b128(                          \
          (gas_v4)(gp + 8), (las_v4)(lp + 8), 0, 0);                           \
    }                                                                          \
  } while (0)
#define ASYNC_WAIT(nv) __builtin_amdgcn_s_wait_asynccnt(nv)
#define ASYNC_FRAG(BPTR, KDIM)                                                 \
  (*(const v16h*)&myB[(size_t)buf * 2048 + nt * 512 + lane * 16])
#else
// fallback: direct global b128 loads into VGPRs
#define ASYNC_DECL __shared__ _Float16 ldsB[1][1][1]
#define ASYNC_ISSUE(K0v, bf) do { } while (0)
#define ASYNC_WAIT(nv)       do { } while (0)
#define ASYNC_FRAG(BPTR, KDIM)                                                 \
  (*(const v16h*)((BPTR) + (size_t)(N0 + nt * 16 + col) * (KDIM) + K0 +        \
                  hi * 16))
#endif

// ---------------------------------------------------------------------------
// Kernel 2: QKV GEMM. qkv[o,l] = w_qkv[o,:] . h[:,l] + b_qkv[o]
// M=1536 (o), N=1024 (l), K=512 (c). One wave -> 16x64 output tile.
// Routes results into q_T / k_T (s>=LE) / v (s>=LE) in concat layout.
// ---------------------------------------------------------------------------
__global__ void __launch_bounds__(128)
qkv_kernel(const float* __restrict__ w, const float* __restrict__ bias,
           const _Float16* __restrict__ hT,
           _Float16* __restrict__ qT, _Float16* __restrict__ kT,
           _Float16* __restrict__ vB) {
  const int b    = blockIdx.z;
  const int wave = threadIdx.x >> 5;
  const int lane = threadIdx.x & 31;
  const int col  = lane & 15;
  const int hi   = lane >> 4;
  const int M0   = blockIdx.y * 64 + wave * 16;
  const int N0   = blockIdx.x * 64;
  const _Float16* Bbase = hT + (size_t)b * L_ * C_;
  constexpr int Kdim_ = C_;
  ASYNC_DECL;

  v8f acc[4] = {};
  GEMM_CORE(C_, acc, w, Bbase);

  #pragma unroll
  for (int nt = 0; nt < 4; nt++) {
    const int l = N0 + nt * 16 + col;
    #pragma unroll
    for (int r = 0; r < 8; r++) {
      const int o   = M0 + r + hi * 8;
      const float v = acc[nt][r] + bias[o];
      const int head = o / (3 * CH_);
      const int r3   = o % (3 * CH_);
      const int bh   = b * H_ + head;
      if (r3 < CH_) {
        qT[((size_t)bh * L_ + l) * CH_ + r3] = (_Float16)(v * SCL_);
      } else if (r3 < 2 * CH_) {
        kT[((size_t)bh * S_ + (LE_ + l)) * CH_ + (r3 - CH_)] =
            (_Float16)(v * SCL_);
      } else {
        vB[((size_t)bh * CH_ + (r3 - 2 * CH_)) * S_ + LE_ + l] = (_Float16)v;
      }
    }
  }
}

// ---------------------------------------------------------------------------
// Kernel 3: EKV GEMM. M=1024 (o), N=128 (le), K=512 (ec).
// Routes ek/ev into columns [0, LE) of k_T / v.
// ---------------------------------------------------------------------------
__global__ void __launch_bounds__(128)
ekv_kernel(const float* __restrict__ w, const float* __restrict__ bias,
           const float* __restrict__ enc,
           _Float16* __restrict__ kT, _Float16* __restrict__ vB) {
  const int b    = blockIdx.z;
  const int wave = threadIdx.x >> 5;
  const int lane = threadIdx.x & 31;
  const int col  = lane & 15;
  const int hi   = lane >> 4;
  const int M0   = blockIdx.y * 64 + wave * 16;
  const int N0   = blockIdx.x * 64;
  const float* eb = enc + (size_t)b * EC_ * LE_;

  v8f acc[4] = {};
  #pragma unroll 2
  for (int K0 = 0; K0 < EC_; K0 += 32) {
    v16h a;
    const float* wr = w + (size_t)(M0 + col) * EC_ + K0 + hi * 16;
    #pragma unroll
    for (int j = 0; j < 8; j++) {
      a[j]     = (_Float16)wr[j];
      a[8 + j] = (_Float16)wr[8 + j];
    }
    #pragma unroll
    for (int nt = 0; nt < 4; nt++) {
      const float* ep = eb + (size_t)(K0 + hi * 16) * LE_ + N0 + nt * 16 + col;
      v16h bb;
      #pragma unroll
      for (int e = 0; e < 16; e++) bb[e] = (_Float16)ep[e * LE_];
      acc[nt] = wmma16(a, bb, acc[nt]);
    }
  }
  #pragma unroll
  for (int nt = 0; nt < 4; nt++) {
    const int l = N0 + nt * 16 + col;   // s index in [0, LE)
    #pragma unroll
    for (int r = 0; r < 8; r++) {
      const int o   = M0 + r + hi * 8;
      const float v = acc[nt][r] + bias[o];
      const int head = o / (2 * CH_);
      const int r2   = o % (2 * CH_);
      const int bh   = b * H_ + head;
      if (r2 < CH_) {
        kT[((size_t)bh * S_ + l) * CH_ + r2] = (_Float16)(v * SCL_);
      } else {
        vB[((size_t)bh * CH_ + (r2 - CH_)) * S_ + l] = (_Float16)v;
      }
    }
  }
}

// ---------------------------------------------------------------------------
// Kernel 4: single-pass flash attention per head.
// One wave handles a 16-row t-tile, all 64 output channels.
// logits = q^T k (2 WMMA per 16-wide s-tile), online softmax, P via LDS
// round-trip (C-layout -> A-layout), 4 PV WMMAs per 32-wide s-chunk.
// ---------------------------------------------------------------------------
__global__ void __launch_bounds__(128)
attn_kernel(const _Float16* __restrict__ qT, const _Float16* __restrict__ kT,
            const _Float16* __restrict__ vB, _Float16* __restrict__ aT) {
  const int bh   = blockIdx.x;           // 0..127
  const int wave = threadIdx.x >> 5;
  const int lane = threadIdx.x & 31;
  const int col  = lane & 15;
  const int hi   = lane >> 4;
  const int T0   = (blockIdx.y * 4 + wave) * 16;

  __shared__ _Float16 ldsP[4][16 * 32];
  _Float16* myP = ldsP[wave];

  // Q fragments for K0 = 0 and 32 (A-layout, contiguous 16-f16 runs)
  v16h aq[2];
  {
    const _Float16* qp = qT + ((size_t)bh * L_ + T0 + col) * CH_;
    #pragma unroll
    for (int kk = 0; kk < 2; kk++) {
      #pragma unroll
      for (int j = 0; j < 8; j++) {
        aq[kk][j]     = qp[kk * 32 + hi * 16 + j];
        aq[kk][8 + j] = qp[kk * 32 + hi * 16 + 8 + j];
      }
    }
  }

  float mrow[8], lsum[8];
  #pragma unroll
  for (int r = 0; r < 8; r++) { mrow[r] = -1e30f; lsum[r] = 0.f; }
  v8f acc[4] = {};

  const _Float16* kb = kT + (size_t)bh * S_ * CH_;
  const _Float16* vb = vB + (size_t)bh * CH_ * S_;

  #pragma unroll 1
  for (int S0 = 0; S0 < S_; S0 += 32) {
    // ---- logits: two 16x16 tiles over s
    v8f d[2] = {};
    #pragma unroll
    for (int half = 0; half < 2; half++) {
      #pragma unroll
      for (int kk = 0; kk < 2; kk++) {
        const _Float16* kp =
            kb + (size_t)(S0 + half * 16 + col) * CH_ + kk * 32 + hi * 16;
        v16h bk = *(const v16h*)kp;       // contiguous 32B
        d[half] = wmma16(aq[kk], bk, d[half]);
      }
    }
    // ---- online softmax update (row stats per lane for its 8 rows)
    #pragma unroll
    for (int r = 0; r < 8; r++) {
      float mx = fmaxf(d[0][r], d[1][r]);
      mx = fmaxf(mx, __shfl_xor(mx, 1, 16));
      mx = fmaxf(mx, __shfl_xor(mx, 2, 16));
      mx = fmaxf(mx, __shfl_xor(mx, 4, 16));
      mx = fmaxf(mx, __shfl_xor(mx, 8, 16));
      const float nm = fmaxf(mrow[r], mx);
      const float sc = __expf(mrow[r] - nm);
      const float e0 = __expf(d[0][r] - nm);
      const float e1 = __expf(d[1][r] - nm);
      float ls = e0 + e1;
      ls += __shfl_xor(ls, 1, 16);
      ls += __shfl_xor(ls, 2, 16);
      ls += __shfl_xor(ls, 4, 16);
      ls += __shfl_xor(ls, 8, 16);
      lsum[r] = lsum[r] * sc + ls;
      mrow[r] = nm;
      #pragma unroll
      for (int ct = 0; ct < 4; ct++) acc[ct][r] *= sc;
      // stash un-normalized P tile (C-layout) in LDS
      const int trow = r + hi * 8;
      myP[trow * 32 + col]      = (_Float16)e0;
      myP[trow * 32 + 16 + col] = (_Float16)e1;
    }
    // ---- reload P in A-layout (in-order per-wave DS ops)
    v16h ap;
    #pragma unroll
    for (int j = 0; j < 8; j++) {
      ap[j]     = myP[col * 32 + hi * 16 + j];
      ap[8 + j] = myP[col * 32 + hi * 16 + 8 + j];
    }
    // ---- PV: acc[t][c] += P[t][s] * v[c][s]
    #pragma unroll
    for (int ct = 0; ct < 4; ct++) {
      const _Float16* vp =
          vb + (size_t)(ct * 16 + col) * S_ + S0 + hi * 16;
      v16h bv = *(const v16h*)vp;         // contiguous 32B
      acc[ct] = wmma16(ap, bv, acc[ct]);
    }
  }

  // ---- normalize and store a_T[b][t][c]  (c global = head*64 + c_local)
  const int b    = bh >> 3;
  const int head = bh & 7;
  #pragma unroll
  for (int ct = 0; ct < 4; ct++) {
    #pragma unroll
    for (int r = 0; r < 8; r++) {
      const float v = acc[ct][r] / lsum[r];
      const int t = T0 + r + hi * 8;
      const int c = head * CH_ + ct * 16 + col;
      aT[((size_t)b * L_ + t) * C_ + c] = (_Float16)v;
    }
  }
}

// ---------------------------------------------------------------------------
// Kernel 5: proj GEMM + bias + residual. M=512, N=1024, K=512. fp32 out.
// ---------------------------------------------------------------------------
__global__ void __launch_bounds__(128)
proj_kernel(const float* __restrict__ w, const float* __restrict__ bias,
            const _Float16* __restrict__ aT, const float* __restrict__ x,
            float* __restrict__ out) {
  const int b    = blockIdx.z;
  const int wave = threadIdx.x >> 5;
  const int lane = threadIdx.x & 31;
  const int col  = lane & 15;
  const int hi   = lane >> 4;
  const int M0   = blockIdx.y * 64 + wave * 16;
  const int N0   = blockIdx.x * 64;
  const _Float16* Bbase = aT + (size_t)b * L_ * C_;
  constexpr int Kdim_ = C_;
  ASYNC_DECL;

  v8f acc[4] = {};
  GEMM_CORE(C_, acc, w, Bbase);

  #pragma unroll
  for (int nt = 0; nt < 4; nt++) {
    const int l = N0 + nt * 16 + col;
    #pragma unroll
    for (int r = 0; r < 8; r++) {
      const int o = M0 + r + hi * 8;
      const size_t idx = ((size_t)b * C_ + o) * L_ + l;
      out[idx] = x[idx] + acc[nt][r] + bias[o];
    }
  }
}

// ---------------------------------------------------------------------------
// Host launcher
// ---------------------------------------------------------------------------
extern "C" void kernel_launch(void* const* d_in, const int* in_sizes, int n_in,
                              void* d_out, int out_size, void* d_ws,
                              size_t ws_size, hipStream_t stream) {
  const float* x     = (const float*)d_in[0];
  const float* enc   = (const float*)d_in[1];
  const float* gns   = (const float*)d_in[2];
  const float* gnb   = (const float*)d_in[3];
  const float* wqkv  = (const float*)d_in[4];
  const float* bqkv  = (const float*)d_in[5];
  const float* wekv  = (const float*)d_in[6];
  const float* bekv  = (const float*)d_in[7];
  const float* wproj = (const float*)d_in[8];
  const float* bproj = (const float*)d_in[9];
  float* out = (float*)d_out;

  // workspace layout (f16): hT | qT | kT | v | aT  -> 84 MiB total
  char* ws = (char*)d_ws;
  const size_t hT_bytes = (size_t)B_ * L_ * C_ * 2;          // 16 MiB
  const size_t qT_bytes = (size_t)B_ * H_ * L_ * CH_ * 2;    // 16 MiB
  const size_t kT_bytes = (size_t)B_ * H_ * S_ * CH_ * 2;    // 18 MiB
  _Float16* hT = (_Float16*)(ws);
  _Float16* qT = (_Float16*)(ws + hT_bytes);
  _Float16* kT = (_Float16*)(ws + hT_bytes + qT_bytes);
  _Float16* vB = (_Float16*)(ws + hT_bytes + qT_bytes + kT_bytes);
  _Float16* aT = (_Float16*)(ws + hT_bytes + qT_bytes + 2 * kT_bytes);

  gn_kernel<<<dim3(B_ * G_), dim3(256), 0, stream>>>(x, gns, gnb, hT);
  qkv_kernel<<<dim3(L_ / 64, (3 * C_) / 64, B_), dim3(128), 0, stream>>>(
      wqkv, bqkv, hT, qT, kT, vB);
  ekv_kernel<<<dim3(LE_ / 64, (2 * C_) / 64, B_), dim3(128), 0, stream>>>(
      wekv, bekv, enc, kT, vB);
  attn_kernel<<<dim3(B_ * H_, L_ / 64), dim3(128), 0, stream>>>(qT, kT, vB, aT);
  proj_kernel<<<dim3(L_ / 64, C_ / 64, B_), dim3(128), 0, stream>>>(
      wproj, bproj, aT, x, out);
}
